// GemmaAttention_73778948211348
// MI455X (gfx1250) — matrix operand order repaired and checked by reading, not compiled, not run on hardware
//
#include <hip/hip_runtime.h>
#include <hip/hip_bf16.h>

typedef __bf16 bf16;
typedef __bf16 v16bf __attribute__((ext_vector_type(16)));
typedef __bf16 v8bf  __attribute__((ext_vector_type(8)));
typedef float  v8f   __attribute__((ext_vector_type(8)));

#define B_   4
#define S_   2048
#define HID_ 2048
#define NH_  8
#define HD_  256

// ---------------------------------------------------------------------------
// WMMA fragment helpers (layouts per CDNA5 ISA §7.12.2, wave32)
// ---------------------------------------------------------------------------

// A-matrix 16x32 bf16: lane<16 -> row=lane, K {k0..k0+7, k0+16..k0+23};
//                      lane>=16 -> row=lane-16, K {k0+8..k0+15, k0+24..k0+31}
__device__ __forceinline__ v16bf load_a_frag(const bf16* base, int lda, int k0) {
  const int lane = threadIdx.x & 31;
  const int r    = lane & 15;
  const int kh   = (lane >> 4) << 3;
  const bf16* p  = base + (size_t)r * lda + k0 + kh;
  v8bf lo = *reinterpret_cast<const v8bf*>(p);
  v8bf hi = *reinterpret_cast<const v8bf*>(p + 16);
  return __builtin_shufflevector(lo, hi, 0,1,2,3,4,5,6,7,8,9,10,11,12,13,14,15);
}

// B-matrix 32x16 bf16 built from a row-major [N,K] "weight" view (B[k][n]=W[n][k]):
// lane<16 -> col=lane, K {k0..k0+7, k0+16..k0+23}; lane>=16 -> col=lane-16, K {k0+8..k0+15, k0+24..k0+31}
__device__ __forceinline__ v16bf load_b_frag(const bf16* base, int ldb, int k0) {
  const int lane = threadIdx.x & 31;
  const int n    = lane & 15;
  const int koff = (lane >> 4) << 3;
  const bf16* p  = base + (size_t)n * ldb + k0 + koff;
  v8bf lo = *reinterpret_cast<const v8bf*>(p);
  v8bf hi = *reinterpret_cast<const v8bf*>(p + 16);
  return __builtin_shufflevector(lo, hi, 0,1,2,3,4,5,6,7,8,9,10,11,12,13,14,15);
}

__device__ __forceinline__ v8f wmma_bf16(v16bf a, v16bf b, v8f c) {
  return __builtin_amdgcn_wmma_f32_16x16x32_bf16(false, a, false, b, (short)0, c,
                                                 false, false);
}

// CDNA5 async global->LDS copy, 16B per active lane (tracked by ASYNCcnt).
__device__ __forceinline__ void async_copy_b128(const bf16* gsrc, bf16* lds_dst) {
  unsigned lds_off = (unsigned)(size_t)lds_dst;  // low 32 bits = LDS offset
  asm volatile("global_load_async_to_lds_b128 %0, %1, off"
               :: "v"(lds_off), "v"(gsrc) : "memory");
}

__device__ __forceinline__ void wait_asynccnt0() {
  asm volatile("s_wait_asynccnt 0" ::: "memory");
}

// ---------------------------------------------------------------------------
// f32 -> bf16 conversion (4 elems/thread; all sizes are multiples of 4)
// ---------------------------------------------------------------------------
__global__ void cvt_f32_to_bf16(const float* __restrict__ in, bf16* __restrict__ out,
                                int n) {
  int i = (blockIdx.x * blockDim.x + threadIdx.x) * 4;
  if (i >= n) return;
  float4 f = *reinterpret_cast<const float4*>(in + i);
  out[i + 0] = (bf16)f.x;
  out[i + 1] = (bf16)f.y;
  out[i + 2] = (bf16)f.z;
  out[i + 3] = (bf16)f.w;
}

// ---------------------------------------------------------------------------
// GEMM: C[M,N] = A[M,K] (bf16) @ W[N,K]^T (bf16), f32 accumulate.
// Each wave computes a 32x32 tile as 2x2 WMMA tiles (A/B frag reuse).
// Weights + activations live in the 192MB L2, so global-fed frags are fine.
// ---------------------------------------------------------------------------
template <bool OUT_BF16>
__global__ void gemm_xt_wmma(const bf16* __restrict__ A, const bf16* __restrict__ W,
                             void* __restrict__ Cv, int M, int N, int K) {
  const int wave   = blockIdx.x * (blockDim.x >> 5) + ((int)threadIdx.x >> 5);
  const int tn_cnt = N >> 5;
  const int tm_cnt = M >> 5;
  if (wave >= tn_cnt * tm_cnt) return;
  const int tm = (wave / tn_cnt) << 5;
  const int tn = (wave % tn_cnt) << 5;

  v8f acc00 = {}, acc01 = {}, acc10 = {}, acc11 = {};
  const bf16* A0 = A + (size_t)tm * K;
  const bf16* A1 = A0 + (size_t)16 * K;
  const bf16* W0 = W + (size_t)tn * K;
  const bf16* W1 = W0 + (size_t)16 * K;

  for (int k0 = 0; k0 < K; k0 += 32) {
    v16bf a0 = load_a_frag(A0, K, k0);
    v16bf a1 = load_a_frag(A1, K, k0);
    v16bf b0 = load_b_frag(W0, K, k0);
    v16bf b1 = load_b_frag(W1, K, k0);
    acc00 = wmma_bf16(a0, b0, acc00);
    acc01 = wmma_bf16(a0, b1, acc01);
    acc10 = wmma_bf16(a1, b0, acc10);
    acc11 = wmma_bf16(a1, b1, acc11);
  }

  const int lane = threadIdx.x & 31;
  const int n    = lane & 15;
  const int m0   = (lane >> 4) << 3;
  if (OUT_BF16) {
    bf16* C = (bf16*)Cv;
    for (int r = 0; r < 8; r++) {
      C[(size_t)(tm +      m0 + r) * N + tn +      n] = (bf16)acc00[r];
      C[(size_t)(tm +      m0 + r) * N + tn + 16 + n] = (bf16)acc01[r];
      C[(size_t)(tm + 16 + m0 + r) * N + tn +      n] = (bf16)acc10[r];
      C[(size_t)(tm + 16 + m0 + r) * N + tn + 16 + n] = (bf16)acc11[r];
    }
  } else {
    float* C = (float*)Cv;
    for (int r = 0; r < 8; r++) {
      C[(size_t)(tm +      m0 + r) * N + tn +      n] = acc00[r];
      C[(size_t)(tm +      m0 + r) * N + tn + 16 + n] = acc01[r];
      C[(size_t)(tm + 16 + m0 + r) * N + tn +      n] = acc10[r];
      C[(size_t)(tm + 16 + m0 + r) * N + tn + 16 + n] = acc11[r];
    }
  }
}

// ---------------------------------------------------------------------------
// RoPE (theta=10000, positions = s). Q: [B*S, NH*HD] -> head-major [B,NH,S,HD]
// ---------------------------------------------------------------------------
#define LOG2_THETA 13.287712379549449f

__global__ void rope_q_kernel(const bf16* __restrict__ Qp, bf16* __restrict__ Qr) {
  int idx = blockIdx.x * blockDim.x + threadIdx.x;  // B*S*NH*128 threads
  int i   = idx & 127;
  int hh  = (idx >> 7) & 7;
  int t2  = idx >> 10;          // b*S + s
  int s   = t2 & (S_ - 1);
  int b   = t2 >> 11;
  const bf16* src = Qp + (size_t)t2 * (NH_ * HD_) + hh * HD_;
  float x0 = (float)src[i];
  float x1 = (float)src[i + 128];
  float inv = exp2f(-(float)i * (LOG2_THETA / 128.0f));
  float fr  = (float)s * inv;
  float sn, c;
  __sincosf(fr, &sn, &c);
  bf16* dst = Qr + (((size_t)(b * NH_ + hh) * S_) + s) * HD_;
  dst[i]       = (bf16)(x0 * c - x1 * sn);
  dst[i + 128] = (bf16)(x1 * c + x0 * sn);
}

__global__ void rope_k_kernel(const bf16* __restrict__ Kp, bf16* __restrict__ Kr) {
  int idx = blockIdx.x * blockDim.x + threadIdx.x;  // B*S*128 threads
  int i   = idx & 127;
  int t2  = idx >> 7;           // b*S + s
  int s   = t2 & (S_ - 1);
  const bf16* src = Kp + (size_t)t2 * HD_;
  float x0 = (float)src[i];
  float x1 = (float)src[i + 128];
  float inv = exp2f(-(float)i * (LOG2_THETA / 128.0f));
  float fr  = (float)s * inv;
  float sn, c;
  __sincosf(fr, &sn, &c);
  bf16* dst = Kr + (size_t)t2 * HD_;
  dst[i]       = (bf16)(x0 * c - x1 * sn);
  dst[i + 128] = (bf16)(x1 * c + x0 * sn);
}

// ---------------------------------------------------------------------------
// Flash attention: one block per (b, h, 128 query rows); 8 waves x 16 rows.
// Q fragments pinned in VGPRs for the whole key loop. K blocks double-buffered
// via async global->LDS copies (ASYNCcnt overlap); V blocks prefetched into
// registers during compute, scattered (transposed) into LDS after the
// consumers-done barrier. Online softmax in exp2 domain.
// LDS: 2*16K (K) + 16K (Vt) + 8K (P) = 56KB static.
// ---------------------------------------------------------------------------
__global__ void __launch_bounds__(256, 1)
flash_attn_kernel(const bf16* __restrict__ Qr, const bf16* __restrict__ Kr,
                  const bf16* __restrict__ Vb, bf16* __restrict__ attnOut) {
  __shared__ __align__(16) bf16 Klds[2][32 * HD_];  // [buf][key][hd]
  __shared__ __align__(16) bf16 Vt[HD_ * 32];       // [hd][key]
  __shared__ __align__(16) bf16 Plds[8 * 16 * 32];

  const int tid  = threadIdx.x;
  const int w    = tid >> 5;
  const int lane = tid & 31;
  const int half = lane >> 4;
  const int nlo  = lane & 15;

  const int blk = blockIdx.x;          // B*NH*(S/128) = 512
  const int tb  = blk & 15;
  const int h   = (blk >> 4) & 7;
  const int b   = blk >> 7;
  const int q0  = tb * 128 + w * 16;

  const bf16* Qbase = Qr + (((size_t)(b * NH_ + h) * S_) + q0) * HD_;
  const bf16* Kg    = Kr + (size_t)b * S_ * HD_;
  const bf16* Vg    = Vb + (size_t)b * S_ * HD_;

  const int vkey = tid & 31;           // V staging role of this thread
  const int vh0  = (tid >> 5) * 32;

  auto stage_k_async = [&](int kb, int buf) {
    const bf16* src = Kg + (size_t)kb * 32 * HD_;
    for (int i = 0; i < 4; i++) {
      int off = (tid + 256 * i) * 8;
      async_copy_b128(src + off, &Klds[buf][off]);
    }
  };
  auto load_v_regs = [&](int kb, v8bf* vr) {
    const bf16* vsrc = Vg + (size_t)(kb * 32 + vkey) * HD_ + vh0;
    for (int i = 0; i < 4; i++)
      vr[i] = *reinterpret_cast<const v8bf*>(vsrc + i * 8);
  };
  auto store_v_trans = [&](const v8bf* vr) {
    for (int i = 0; i < 4; i++)
      for (int j = 0; j < 8; j++)
        Vt[(vh0 + i * 8 + j) * 32 + vkey] = vr[i][j];
  };

  // Pin the wave's 16x256 Q tile in registers (8 A-fragments, 64 VGPRs):
  // loaded once, reused for every key block.
  v16bf qf[8];
  for (int kk = 0; kk < 8; kk++) qf[kk] = load_a_frag(Qbase, HD_, kk * 32);

  v8f o[16] = {};
  float m[8], l[8];
  for (int r = 0; r < 8; r++) { m[r] = -__builtin_inff(); l[r] = 0.0f; }

  const float SC = 0.09016844005556021f;  // (1/sqrt(256)) * log2(e)
  const int nkb = tb * 4 + 4;             // causal: highest key block needed

  // prologue: stage block 0
  stage_k_async(0, 0);
  {
    v8bf vr[4];
    load_v_regs(0, vr);
    store_v_trans(vr);
  }
  wait_asynccnt0();
  __syncthreads();

  for (int kb = 0; kb < nkb; kb++) {
    const int cur  = kb & 1;
    const bool more = (kb + 1 < nkb);
    v8bf vr[4];
    if (more) {
      stage_k_async(kb + 1, cur ^ 1);  // async: overlaps the WMMAs below
      load_v_regs(kb + 1, vr);         // latency hidden behind compute
    }

    // scores S[16 x 32] = Q_tile @ K_block^T  (two 16x16 C tiles)
    v8f s0 = {}, s1 = {};
    for (int kk = 0; kk < 8; kk++) {
      v16bf bb0 = load_b_frag(&Klds[cur][0], HD_, kk * 32);
      v16bf bb1 = load_b_frag(&Klds[cur][16 * HD_], HD_, kk * 32);
      s0 = wmma_bf16(qf[kk], bb0, s0);
      s1 = wmma_bf16(qf[kk], bb1, s1);
    }

    // scale (into exp2 domain) + causal mask
    for (int r = 0; r < 8; r++) {
      int q   = q0 + r + 8 * half;
      int k0g = kb * 32 + nlo;
      s0[r] = (k0g      <= q) ? s0[r] * SC : -__builtin_inff();
      s1[r] = (k0g + 16 <= q) ? s1[r] * SC : -__builtin_inff();
    }

    // online softmax: row reductions across each 16-lane half (wave32)
    float alpha[8];
    for (int r = 0; r < 8; r++) {
      float v = fmaxf(s0[r], s1[r]);
      v = fmaxf(v, __shfl_xor(v, 1));
      v = fmaxf(v, __shfl_xor(v, 2));
      v = fmaxf(v, __shfl_xor(v, 4));
      v = fmaxf(v, __shfl_xor(v, 8));
      float mn = fmaxf(m[r], v);
      alpha[r] = exp2f(m[r] - mn);
      m[r] = mn;
      float p0 = exp2f(s0[r] - mn);
      float p1 = exp2f(s1[r] - mn);
      s0[r] = p0;
      s1[r] = p1;
      float rs = p0 + p1;
      rs += __shfl_xor(rs, 1);
      rs += __shfl_xor(rs, 2);
      rs += __shfl_xor(rs, 4);
      rs += __shfl_xor(rs, 8);
      l[r] = l[r] * alpha[r] + rs;
    }
    for (int t = 0; t < 16; t++)
      for (int r = 0; r < 8; r++) o[t][r] *= alpha[r];

    // P (C layout) -> LDS -> A-fragment layout; explicit DS wait for the
    // cross-lane LDS dependency (same wave, lockstep, compiler can't see it)
    bf16* Pw = &Plds[w * 512];
    for (int r = 0; r < 8; r++) {
      Pw[(r + 8 * half) * 32 + nlo]      = (bf16)s0[r];
      Pw[(r + 8 * half) * 32 + 16 + nlo] = (bf16)s1[r];
    }
    asm volatile("s_wait_dscnt 0" ::: "memory");
    v16bf ap = load_a_frag(Pw, 32, 0);

    // O[16 x 256] += P[16 x 32] @ V[32 x 256]
    for (int t = 0; t < 16; t++) {
      v16bf bv = load_b_frag(&Vt[t * 16 * 32], 32, 0);
      o[t] = wmma_bf16(ap, bv, o[t]);
    }

    __syncthreads();        // all waves done reading Vt (and Klds[cur])
    if (more) store_v_trans(vr);
    wait_asynccnt0();       // K block kb+1 resident in Klds[cur^1]
    __syncthreads();        // Vt + Klds[cur^1] published
  }

  // epilogue: O / l -> bf16, layout [b*S + s, h*HD + hd]
  for (int r = 0; r < 8; r++) l[r] = 1.0f / l[r];
  for (int t = 0; t < 16; t++) {
    for (int r = 0; r < 8; r++) {
      int srow = q0 + r + 8 * half;
      int col  = h * HD_ + t * 16 + nlo;
      attnOut[(size_t)(b * S_ + srow) * (NH_ * HD_) + col] = (bf16)(o[t][r] * l[r]);
    }
  }
}

// ---------------------------------------------------------------------------
// Launcher
// ---------------------------------------------------------------------------
extern "C" void kernel_launch(void* const* d_in, const int* in_sizes, int n_in,
                              void* d_out, int out_size, void* d_ws, size_t ws_size,
                              hipStream_t stream) {
  (void)in_sizes; (void)n_in; (void)out_size; (void)ws_size;
  const float* hs = (const float*)d_in[0];
  // d_in[1] = attention_mask (pure causal -> computed inline), d_in[2] = position_ids
  const float* Wq = (const float*)d_in[3];
  const float* Wk = (const float*)d_in[4];
  const float* Wv = (const float*)d_in[5];
  const float* Wo = (const float*)d_in[6];

  char* ws = (char*)d_ws;
  size_t off = 0;
  auto alloc = [&](size_t bytes) {
    void* p = ws + off;
    off += (bytes + 255) & ~(size_t)255;
    return p;
  };
  const size_t M = (size_t)B_ * S_;  // 8192
  bf16* Xb  = (bf16*)alloc(M * HID_ * 2);
  bf16* Wqb = (bf16*)alloc((size_t)HID_ * HID_ * 2);
  bf16* Wkb = (bf16*)alloc((size_t)HD_ * HID_ * 2);
  bf16* Wvb = (bf16*)alloc((size_t)HD_ * HID_ * 2);
  bf16* Wob = (bf16*)alloc((size_t)HID_ * HID_ * 2);
  bf16* Qp  = (bf16*)alloc(M * (NH_ * HD_) * 2);
  bf16* Kp  = (bf16*)alloc(M * HD_ * 2);
  bf16* Vbf = (bf16*)alloc(M * HD_ * 2);
  bf16* Qr  = (bf16*)alloc(M * (NH_ * HD_) * 2);
  bf16* Kr  = (bf16*)alloc(M * HD_ * 2);
  bf16* At  = (bf16*)alloc(M * (NH_ * HD_) * 2);

  auto cvt = [&](const float* src, bf16* dst, size_t n) {
    int grid = (int)((n / 4 + 255) / 256);
    cvt_f32_to_bf16<<<grid, 256, 0, stream>>>(src, dst, (int)n);
  };
  cvt(hs, Xb, M * HID_);
  cvt(Wq, Wqb, (size_t)HID_ * HID_);
  cvt(Wk, Wkb, (size_t)HD_ * HID_);
  cvt(Wv, Wvb, (size_t)HD_ * HID_);
  cvt(Wo, Wob, (size_t)HID_ * HID_);

  auto gemm_bf = [&](const bf16* A, const bf16* W, bf16* C, int Mm, int Nn, int Kk) {
    int waves = (Mm / 32) * (Nn / 32);
    gemm_xt_wmma<true><<<(waves + 7) / 8, 256, 0, stream>>>(A, W, (void*)C, Mm, Nn, Kk);
  };
  gemm_bf(Xb, Wqb, Qp, (int)M, NH_ * HD_, HID_);  // Q
  gemm_bf(Xb, Wkb, Kp, (int)M, HD_, HID_);        // K
  gemm_bf(Xb, Wvb, Vbf, (int)M, HD_, HID_);       // V

  rope_q_kernel<<<(B_ * S_ * NH_ * 128) / 256, 256, 0, stream>>>(Qp, Qr);
  rope_k_kernel<<<(B_ * S_ * 128) / 256, 256, 0, stream>>>(Kp, Kr);

  flash_attn_kernel<<<B_ * NH_ * (S_ / 128), 256, 0, stream>>>(Qr, Kr, Vbf, At);

  {  // out = attn @ Wo^T, f32 output
    int waves = ((int)M / 32) * (HID_ / 32);
    gemm_xt_wmma<false><<<(waves + 7) / 8, 256, 0, stream>>>(At, Wob, d_out,
                                                             (int)M, HID_, HID_);
  }
}
